// NearestKProjector_77988016161038
// MI455X (gfx1250) — matrix operand order repaired and checked by reading, not compiled, not run on hardware
//
#include <hip/hip_runtime.h>
#include <hip/hip_bf16.h>

typedef __attribute__((ext_vector_type(16))) _Float16 v16h;
typedef __attribute__((ext_vector_type(8)))  _Float16 v8h;
typedef __attribute__((ext_vector_type(8)))  float    v8f;
typedef __attribute__((ext_vector_type(4)))  float    v4f;

#define DIM    2048
#define VOCAB  32000
#define NROWS  8192
#define KTOP   5
#define MT     32             // M-tile rows per workgroup
#define NTILES (VOCAB / 16)   // 2000 column tiles
#define NGROUP (NTILES / 16)  // 125 groups of 16 tiles (2 per wave)
#define ALPHA  0.5f
#define TEMPR  10.0f

// ---------------------------------------------------------------------------
// Kernel 1: L2-normalize rows (f32 in) -> f16 out. One block per row.
// ---------------------------------------------------------------------------
__global__ __launch_bounds__(256) void l2norm_f16_kernel(
    const float* __restrict__ in, _Float16* __restrict__ out) {
  const int row = blockIdx.x;
  const float* r = in + (size_t)row * DIM;
  _Float16* o   = out + (size_t)row * DIM;

  float ss = 0.f;
  for (int c = threadIdx.x; c < DIM; c += 256) {
    float v = r[c];
    ss += v * v;
  }
  __shared__ float red[256];
  red[threadIdx.x] = ss;
  __syncthreads();
  for (int s = 128; s > 0; s >>= 1) {
    if (threadIdx.x < s) red[threadIdx.x] += red[threadIdx.x + s];
    __syncthreads();
  }
  const float inv = 1.0f / fmaxf(sqrtf(red[0]), 1e-12f);
  for (int c = threadIdx.x; c < DIM; c += 256) {
    o[c] = (_Float16)(r[c] * inv);
  }
}

// ---------------------------------------------------------------------------
// Kernel 2: fused sim-GEMM (WMMA f16) + running top-5 + softmax + projection.
// 8 waves / 32 rows per workgroup; each wave: 2 M-subtiles x 2 N-tiles
// => 4 WMMAs per k-step with 1 ds_b128 + 1 global_b128 per WMMA.
// A-tile rows are sub-tile interleaved in LDS: As[m][0][*]=row m,
// As[m][1][*]=row m+16, so every A access is a 16-bit DS immediate
// off one base register (no per-iteration address VALU, no hazard NOPs).
// ---------------------------------------------------------------------------
__global__ __launch_bounds__(256) void nkproj_kernel(
    const float*    __restrict__ emb,    // [NROWS, DIM] f32 (original)
    const float*    __restrict__ voc,    // [VOCAB, DIM] f32 (original)
    const _Float16* __restrict__ embh,   // [NROWS, DIM] f16 normalized
    const _Float16* __restrict__ voch,   // [VOCAB, DIM] f16 normalized
    float*          __restrict__ out) {  // [NROWS, DIM] f32

  __shared__ _Float16 As[16][2][DIM];       // 128 KB  interleaved A-tile
  __shared__ float    simb[8][MT][16];      // 16 KB   per-wave sim tile buffer
  __shared__ float    msim[8][MT][KTOP];    // 5 KB    per-wave top-5 sims
  __shared__ int      midx[8][MT][KTOP];    // 5 KB    per-wave top-5 indices
  __shared__ float    wgt[MT][KTOP];        // softmax weights
  __shared__ int      wix[MT][KTOP];        // final indices

  const int tid  = threadIdx.x;
  const int wave = tid >> 5;
  const int lane = tid & 31;
  const int row0 = blockIdx.x * MT;

  // ---- stage A tile (32 rows), interleaving subtiles: row r -> As[r&15][r>>4]
  {
    const v8h* src = (const v8h*)(embh + (size_t)row0 * DIM);
    v8h* dst = (v8h*)(&As[0][0][0]);
    const int nv = MT * DIM / 8;  // 8192 16-byte vectors
    for (int i = tid; i < nv; i += 256) {
      const int r    = i >> 8;        // source row (0..31)
      const int col8 = i & 255;       // v8h column within row
      const int rr   = ((r & 15) << 1) | (r >> 4);
      dst[(rr << 8) + col8] = src[i];
    }
  }
  __syncthreads();

  // per-lane running top-5: row = lane (all 32 lanes active)
  float ts[KTOP];
  int   tix[KTOP];
#pragma unroll
  for (int k = 0; k < KTOP; ++k) { ts[k] = -1e30f; tix[k] = 0; }

  const int mrow = lane & 15;
  const int kA = (lane >> 4) << 4;   // A: 0 or 16  (contiguous 16 K per lane-half)
  const int kB = (lane >> 4) << 4;   // B: 0 or 16  (contiguous 16 K per lane-half)

  // Single LDS base: subtile 0 at +0, subtile 1 at +4KB (256 v8h), all
  // 64 k-steps within the 16-bit DS immediate range.
  const v8h* ap = (const v8h*)&As[mrow][0][kA];

  // ---- 125 groups of 16 column tiles; wave w owns tiles g*16+2w, g*16+2w+1 --
  for (int g = 0; g < NGROUP; ++g) {
    const int ct0 = g * 16 + wave * 2;
    v8f acc00 = {}, acc01 = {};   // rows 0..15  x tiles {ct0, ct0+1}
    v8f acc10 = {}, acc11 = {};   // rows 16..31 x tiles {ct0, ct0+1}

    // Both B tiles hang off one base: second tile is +16*DIM halves = +64KB,
    // within the 24-bit global immediate range.
    const _Float16* bbase = voch + (size_t)(ct0 * 16 + mrow) * DIM + kB;

    if (g + 1 < NGROUP) {  // prefetch next group's first tile (L2-resident)
      __builtin_prefetch(bbase + (size_t)16 * 16 * DIM, 0, 1);
    }

#pragma unroll 2
    for (int ks = 0; ks < DIM / 32; ++ks) {
      const int k0 = ks * 32;
      // B fragments (global/L2): 32B each, immediate offsets off bbase
      v16h b0 = *(const v16h*)(bbase + k0);
      v16h b1 = *(const v16h*)(bbase + 16 * DIM + k0);
      // A fragments (LDS): immediate offsets off ap
      v8h a0lo = ap[ks * 4];
      v8h a0hi = ap[ks * 4 + 1];
      v8h a1lo = ap[ks * 4 + 256];      // +4096 bytes: subtile 1
      v8h a1hi = ap[ks * 4 + 257];
      v16h a0 = __builtin_shufflevector(a0lo, a0hi,
          0, 1, 2, 3, 4, 5, 6, 7, 8, 9, 10, 11, 12, 13, 14, 15);
      v16h a1 = __builtin_shufflevector(a1lo, a1hi,
          0, 1, 2, 3, 4, 5, 6, 7, 8, 9, 10, 11, 12, 13, 14, 15);
      acc00 = __builtin_amdgcn_wmma_f32_16x16x32_f16(
          false, a0, false, b0, (short)0, acc00, false, false);
      acc10 = __builtin_amdgcn_wmma_f32_16x16x32_f16(
          false, a1, false, b0, (short)0, acc10, false, false);
      acc01 = __builtin_amdgcn_wmma_f32_16x16x32_f16(
          false, a0, false, b1, (short)0, acc01, false, false);
      acc11 = __builtin_amdgcn_wmma_f32_16x16x32_f16(
          false, a1, false, b1, (short)0, acc11, false, false);
    }

    // Spill + scan is wave-private: LDS ops from one wave execute in order
    // (DScnt), so s_wait_dscnt 0 + compiler fence replaces __syncthreads.
    const int rbase = (lane >> 4) << 3;
    const v4f* srow = (const v4f*)&simb[wave][lane][0];

    // ---- round 0: tile ct0 ----
#pragma unroll
    for (int r = 0; r < 8; ++r) simb[wave][rbase + r][mrow]      = acc00[r];
#pragma unroll
    for (int r = 0; r < 8; ++r) simb[wave][16 + rbase + r][mrow] = acc10[r];
    asm volatile("s_wait_dscnt 0x0" ::: "memory");
    {
      const int cbase = ct0 * 16;
#pragma unroll
      for (int cc = 0; cc < 4; ++cc) {
        v4f sv = srow[cc];
#pragma unroll
        for (int j = 0; j < 4; ++j) {
          const float s = sv[j];
          if (s > ts[KTOP - 1]) {
            int p = KTOP - 1;
#pragma unroll
            for (int q = KTOP - 1; q > 0; --q) {
              if (ts[q - 1] < s) { ts[q] = ts[q - 1]; tix[q] = tix[q - 1]; p = q - 1; }
            }
            ts[p] = s; tix[p] = cbase + cc * 4 + j;
          }
        }
      }
    }
    asm volatile("" ::: "memory");  // keep round-1 stores after round-0 loads

    // ---- round 1: tile ct0+1 ----
#pragma unroll
    for (int r = 0; r < 8; ++r) simb[wave][rbase + r][mrow]      = acc01[r];
#pragma unroll
    for (int r = 0; r < 8; ++r) simb[wave][16 + rbase + r][mrow] = acc11[r];
    asm volatile("s_wait_dscnt 0x0" ::: "memory");
    {
      const int cbase = (ct0 + 1) * 16;
#pragma unroll
      for (int cc = 0; cc < 4; ++cc) {
        v4f sv = srow[cc];
#pragma unroll
        for (int j = 0; j < 4; ++j) {
          const float s = sv[j];
          if (s > ts[KTOP - 1]) {
            int p = KTOP - 1;
#pragma unroll
            for (int q = KTOP - 1; q > 0; --q) {
              if (ts[q - 1] < s) { ts[q] = ts[q - 1]; tix[q] = tix[q - 1]; p = q - 1; }
            }
            ts[p] = s; tix[p] = cbase + cc * 4 + j;
          }
        }
      }
    }
    asm volatile("" ::: "memory");
  }

  // ---- merge the 8 per-wave top-5 lists ----
#pragma unroll
  for (int k = 0; k < KTOP; ++k) {
    msim[wave][lane][k] = ts[k];
    midx[wave][lane][k] = tix[k];
  }
  __syncthreads();

  if (tid < MT) {
    float bs[KTOP]; int bi[KTOP];
#pragma unroll
    for (int k = 0; k < KTOP; ++k) { bs[k] = -1e30f; bi[k] = 0; }
    for (int w = 0; w < 8; ++w) {
#pragma unroll
      for (int k = 0; k < KTOP; ++k) {
        const float s = msim[w][tid][k];
        const int  vi = midx[w][tid][k];
        if (s > bs[KTOP - 1]) {
          int p = KTOP - 1;
#pragma unroll
          for (int q = KTOP - 1; q > 0; --q) {
            if (bs[q - 1] < s) { bs[q] = bs[q - 1]; bi[q] = bi[q - 1]; p = q - 1; }
          }
          bs[p] = s; bi[p] = vi;
        }
      }
    }
    // softmax over top-5 with temperature
    const float mx = bs[0];
    float e[KTOP], sum = 0.f;
#pragma unroll
    for (int k = 0; k < KTOP; ++k) { e[k] = __expf((bs[k] - mx) * TEMPR); sum += e[k]; }
    const float inv = 1.0f / sum;
#pragma unroll
    for (int k = 0; k < KTOP; ++k) { wgt[tid][k] = e[k] * inv; wix[tid][k] = bi[k]; }
  }
  __syncthreads();

  // ---- projection: out = ALPHA*emb + (1-ALPHA) * sum_k w_k * vocab[idx_k] ----
  for (int r = 0; r < MT; ++r) {
    const float* er = emb + (size_t)(row0 + r) * DIM;
    float* orow = out + (size_t)(row0 + r) * DIM;
    float wk[KTOP];
    const float* vr[KTOP];
#pragma unroll
    for (int k = 0; k < KTOP; ++k) {
      wk[k] = wgt[r][k];
      vr[k] = voc + (size_t)wix[r][k] * DIM;
    }
    for (int c = tid; c < DIM; c += 256) {
      float p = 0.f;
#pragma unroll
      for (int k = 0; k < KTOP; ++k) p += wk[k] * vr[k][c];
      orow[c] = ALPHA * er[c] + (1.0f - ALPHA) * p;
    }
  }
}

// ---------------------------------------------------------------------------
extern "C" void kernel_launch(void* const* d_in, const int* in_sizes, int n_in,
                              void* d_out, int out_size, void* d_ws, size_t ws_size,
                              hipStream_t stream) {
  const float* emb = (const float*)d_in[0];   // [4,2048,2048] -> [8192,2048]
  const float* voc = (const float*)d_in[1];   // [32000,2048]
  float* out = (float*)d_out;

  _Float16* embh = (_Float16*)d_ws;                       // 33.5 MB
  _Float16* voch = embh + (size_t)NROWS * DIM;            // 131  MB (fits L2)

  l2norm_f16_kernel<<<NROWS, 256, 0, stream>>>(emb, embh);
  l2norm_f16_kernel<<<VOCAB, 256, 0, stream>>>(voc, voch);
  nkproj_kernel<<<NROWS / MT, 256, 0, stream>>>(emb, voc, embh, voch, out);
}
